// DolphinPointwiseFFN2d_60035052863749
// MI455X (gfx1250) — matrix operand order repaired and checked by reading, not compiled, not run on hardware
//
#include <hip/hip_runtime.h>
#include <hip/hip_bf16.h>

// DolphinPointwiseFFN2d for MI455X (gfx1250, wave32, WMMA f32_16x16x32_f16)
//
// B=4, C=128, F=128, T=512, H=256.  Two-pass fused pipeline:
//   k1: RMSNorm + GEMM1(512x128 @ 128xQ) + GLU  -> v (f16, in d_ws)
//   k2: depthwise 3x3 (causal T) + SiLU + GEMM2(128x256 @ 256xQ) + bias + residual

typedef _Float16 v16h __attribute__((ext_vector_type(16)));
typedef _Float16 v8h  __attribute__((ext_vector_type(8)));
typedef float    v8f  __attribute__((ext_vector_type(8)));

#define CB 4
#define CC 128
#define CF 128
#define CT 512
#define CH 256
#define FT (CF * CT)   // 65536 pixels per (b, channel-plane)

__device__ __forceinline__ float sigmoidf_(float x) {
    return 1.0f / (1.0f + __expf(-x));
}

// ---- WMMA fragment loaders (CDNA5 ISA 7.12.2 layouts, wave32) ----

// A fragment: 16x32 f16 (MxK), source row-major with leading dim ldk (halves).
// lane m = l&15 ; hi = l>>4 ; elems 0..7 -> K = 8*hi + e ; elems 8..15 -> K = 16 + 8*hi + (e-8)
__device__ __forceinline__ v16h load_a_frag(const _Float16* __restrict__ base,
                                            int row0, int ldk, int k0, int lane) {
    int m  = lane & 15;
    int hi = lane >> 4;
    const _Float16* p = base + (size_t)(row0 + m) * ldk + k0;
    v8h lo = *(const v8h*)(p + 8 * hi);
    v8h hh = *(const v8h*)(p + 16 + 8 * hi);
    v16h r;
#pragma unroll
    for (int i = 0; i < 8; ++i) { r[i] = lo[i]; r[8 + i] = hh[i]; }
    return r;
}

// B fragment: 32x16 f16 (KxN), source stored PIXEL-major in LDS: [n][k], stride ldk halves.
// lane n = l&15 ; hi = l>>4 ; elems 0..7 -> K = 8*hi + e ; elems 8..15 -> K = 16 + 8*hi + (e-8)
__device__ __forceinline__ v16h load_b_frag(const _Float16* base,
                                            int n0, int ldk, int k0, int lane) {
    int n  = lane & 15;
    int hi = lane >> 4;
    const _Float16* p = base + (size_t)(n0 + n) * ldk + k0;
    v8h lo = *(const v8h*)(p + 8 * hi);
    v8h hh = *(const v8h*)(p + 16 + 8 * hi);
    v16h r;
#pragma unroll
    for (int i = 0; i < 8; ++i) { r[i] = lo[i]; r[8 + i] = hh[i]; }
    return r;
}

// ---- k0: weight conversion f32 -> f16 into workspace ----
__global__ void k0_prep(const float* __restrict__ w_in, const float* __restrict__ w_out,
                        _Float16* __restrict__ w_in_h, _Float16* __restrict__ w_out_h) {
    int i = blockIdx.x * 256 + threadIdx.x;
    if (i < 2 * CH * CC) w_in_h[i] = (_Float16)w_in[i];     // 512*128
    if (i < CC * CH)     w_out_h[i] = (_Float16)w_out[i];   // 128*256
}

// ---- k1: RMSNorm + GEMM1 + GLU ----
// grid: CB * (FT/32) = 8192 blocks, 256 threads (8 waves).
// Block computes a 512 x 32 ab-tile; wave w owns GLU-paired rows
// {w*32 + [0,32)} (a-part) and {256 + w*32 + [0,32)} (b-part).
#define K1_LDY 136   // halves; 272B row: 16B aligned, bank stride 68 dwords
__global__ __launch_bounds__(256) void k1_gemm1_glu(
    const float* __restrict__ x, const float* __restrict__ g_norm,
    const float* __restrict__ b_in, const _Float16* __restrict__ w_in_h,
    _Float16* __restrict__ v) {
    __shared__ _Float16 yls[32 * K1_LDY];   // normalized f16 tile, pixel-major [q][c]
    __shared__ float psum[8][32];
    __shared__ float scale[32];
    __shared__ float gls[CC];
    __shared__ float bls[2 * CH];

    const int tid  = threadIdx.x;
    const int lane = tid & 31;
    const int wave = tid >> 5;
    const int blk  = blockIdx.x;
    const int b    = blk >> 11;              // 2048 tiles per batch
    const int q0   = (blk & 2047) * 32;      // flat pixel index within batch plane

    if (tid < CC) gls[tid] = g_norm[tid];
    bls[tid]       = b_in[tid];
    bls[tid + 256] = b_in[tid + 256];

    // Load x slab: thread (cg=tid>>5, q=tid&31) holds channels cg*16..cg*16+15 of pixel q0+q.
    const int q  = tid & 31;
    const int cg = tid >> 5;
    const float* xb = x + (size_t)b * CC * FT + q0 + q;
    float xv[16];
    float ss = 0.f;
#pragma unroll
    for (int i = 0; i < 16; ++i) {
        float val = xb[(size_t)(cg * 16 + i) * FT];
        xv[i] = val;
        ss += val * val;
    }
    psum[cg][q] = ss;
    __syncthreads();
    if (tid < 32) {
        float s = 0.f;
#pragma unroll
        for (int j = 0; j < 8; ++j) s += psum[j][tid];
        scale[tid] = rsqrtf(s * (1.0f / CC) + 1e-6f);
    }
    __syncthreads();
    {
        const float sc = scale[q];
#pragma unroll
        for (int i = 0; i < 16; ++i) {
            int c = cg * 16 + i;
            yls[q * K1_LDY + c] = (_Float16)(xv[i] * sc * gls[c]);
        }
    }
    __syncthreads();

    // GEMM1: A = w_in_h (512x128), B = yls (128x32), K in 4 steps of 32.
    const int oA = wave * 32;
    v8f zero_ = {};
    v8f acc[4][2];
#pragma unroll
    for (int mt = 0; mt < 4; ++mt) { acc[mt][0] = zero_; acc[mt][1] = zero_; }

#pragma unroll
    for (int kk = 0; kk < 4; ++kk) {
        const int k0 = kk * 32;
        v16h bf0 = load_b_frag(yls, 0,  K1_LDY, k0, lane);
        v16h bf1 = load_b_frag(yls, 16, K1_LDY, k0, lane);
#pragma unroll
        for (int mt = 0; mt < 4; ++mt) {
            int row0 = (mt < 2) ? (oA + mt * 16) : (CH + oA + (mt - 2) * 16);
            v16h af = load_a_frag(w_in_h, row0, CC, k0, lane);
            acc[mt][0] = __builtin_amdgcn_wmma_f32_16x16x32_f16(
                false, af, false, bf0, (short)0, acc[mt][0], false, false);
            acc[mt][1] = __builtin_amdgcn_wmma_f32_16x16x32_f16(
                false, af, false, bf1, (short)0, acc[mt][1], false, false);
        }
    }

    // GLU in registers: a-part tile mt pairs with b-part tile mt+2 (same m,n element-wise).
    const int hi = lane >> 4;
    const int nn = lane & 15;
    _Float16* vb = v + (size_t)b * CH * FT;
#pragma unroll
    for (int mt = 0; mt < 2; ++mt) {
#pragma unroll
        for (int nt = 0; nt < 2; ++nt) {
            const int qq = q0 + nt * 16 + nn;
#pragma unroll
            for (int r = 0; r < 8; ++r) {
                int h = oA + mt * 16 + r + 8 * hi;
                float av = acc[mt][nt][r] + bls[h];
                float bv = acc[mt + 2][nt][r] + bls[h + CH];
                vb[(size_t)h * FT + qq] = (_Float16)(av * sigmoidf_(bv));
            }
        }
    }
}

// ---- k2: depthwise 3x3 (F pad (1,1), causal T pad (2,0)) + SiLU + GEMM2 + residual ----
// grid: CB * CF * (CT/32) = 8192 blocks, 256 threads (8 waves).
#define K2_LDS 264   // halves; 528B row: 16B aligned, bank stride 132 dwords
__global__ __launch_bounds__(256) void k2_dwconv_gemm2(
    const float* __restrict__ x, const float* __restrict__ w_dw,
    const float* __restrict__ b_dw, const float* __restrict__ b_out,
    const _Float16* __restrict__ w_out_h, const _Float16* __restrict__ v,
    float* __restrict__ out) {
    __shared__ _Float16 sls[32 * K2_LDS];   // SiLU(conv) tile, pixel-major [q][h]
    __shared__ float wdw[CH][9];
    __shared__ float bdw[CH];
    __shared__ float bout[CC];

    const int tid  = threadIdx.x;
    const int lane = tid & 31;
    const int wave = tid >> 5;
    const int blk  = blockIdx.x;
    const int b    = blk >> 11;
    const int rem  = blk & 2047;
    const int f    = rem >> 4;
    const int t0   = (rem & 15) * 32;

    {
        int h = tid;
#pragma unroll
        for (int k = 0; k < 9; ++k) wdw[h][k] = w_dw[h * 9 + k];
        bdw[h] = b_dw[h];
        if (tid < CC) bout[tid] = b_out[tid];
    }
    __syncthreads();

    // Depthwise conv: thread (tx = T-lane, ty) covers h = {ty, ty+8, ...}; lanes coalesce on T.
    const int tx = tid & 31;
    const int ty = tid >> 5;
    const _Float16* vb = v + (size_t)b * CH * FT;
#pragma unroll 4
    for (int hh = 0; hh < 32; ++hh) {
        const int h = hh * 8 + ty;
        float acc = 0.f;
        const _Float16* vh = vb + (size_t)h * FT;
#pragma unroll
        for (int df = 0; df < 3; ++df) {
            const int fy = f - 1 + df;
            if ((unsigned)fy < (unsigned)CF) {
                const _Float16* vr = vh + (size_t)fy * CT + (t0 + tx - 2);
#pragma unroll
                for (int dt = 0; dt < 3; ++dt) {
                    const int tt = t0 + tx - 2 + dt;     // causal: never exceeds T-1
                    float vv = (tt >= 0) ? (float)vr[dt] : 0.f;
                    acc += wdw[h][df * 3 + dt] * vv;
                }
            }
        }
        acc += bdw[h];
        sls[tx * K2_LDS + h] = (_Float16)(acc * sigmoidf_(acc));
    }
    __syncthreads();

    // GEMM2: A = w_out_h (128x256), B = sls (256x32); wave w -> output rows c = 16w..16w+15.
    v8f zero_ = {};
    v8f acc2[2] = {zero_, zero_};
#pragma unroll
    for (int kk = 0; kk < 8; ++kk) {
        const int k0 = kk * 32;
        v16h af  = load_a_frag(w_out_h, wave * 16, CH, k0, lane);
        v16h bf0 = load_b_frag(sls, 0,  K2_LDS, k0, lane);
        v16h bf1 = load_b_frag(sls, 16, K2_LDS, k0, lane);
        acc2[0] = __builtin_amdgcn_wmma_f32_16x16x32_f16(
            false, af, false, bf0, (short)0, acc2[0], false, false);
        acc2[1] = __builtin_amdgcn_wmma_f32_16x16x32_f16(
            false, af, false, bf1, (short)0, acc2[1], false, false);
    }

    const int hi = lane >> 4;
    const int nn = lane & 15;
#pragma unroll
    for (int nt = 0; nt < 2; ++nt) {
        const int t = t0 + nt * 16 + nn;
#pragma unroll
        for (int r = 0; r < 8; ++r) {
            const int c = wave * 16 + r + 8 * hi;
            const size_t idx = (((size_t)b * CC + c) * CF + f) * CT + t;
            out[idx] = x[idx] + bout[c] + acc2[nt][r];
        }
    }
}

extern "C" void kernel_launch(void* const* d_in, const int* in_sizes, int n_in,
                              void* d_out, int out_size, void* d_ws, size_t ws_size,
                              hipStream_t stream) {
    (void)in_sizes; (void)n_in; (void)out_size; (void)ws_size;
    const float* x      = (const float*)d_in[0];
    const float* g_norm = (const float*)d_in[1];
    const float* w_in   = (const float*)d_in[2];
    const float* b_in   = (const float*)d_in[3];
    const float* w_dw   = (const float*)d_in[4];
    const float* b_dw   = (const float*)d_in[5];
    const float* w_out  = (const float*)d_in[6];
    const float* b_out  = (const float*)d_in[7];
    float* out = (float*)d_out;

    // Workspace layout (f16): w_in_h[65536] | w_out_h[32768] | v[4*256*65536]
    _Float16* w_in_h  = (_Float16*)d_ws;
    _Float16* w_out_h = w_in_h + 2 * CH * CC;
    _Float16* v       = w_out_h + CC * CH;

    k0_prep<<<256, 256, 0, stream>>>(w_in, w_out, w_in_h, w_out_h);
    k1_gemm1_glu<<<CB * (FT / 32), 256, 0, stream>>>(x, g_norm, b_in, w_in_h, v);
    k2_dwconv_gemm2<<<CB * CF * (CT / 32), 256, 0, stream>>>(
        x, w_dw, b_dw, b_out, w_out_h, v, out);
}
